// GaussianHierarchicalDownsampler_23819888623884
// MI455X (gfx1250) — matrix operand order repaired and checked by reading, not compiled, not run on hardware
//
#include <hip/hip_runtime.h>
#include <math.h>
#include <stdint.h>

// ---------------------------------------------------------------- constants
constexpr int   B_  = 2;
constexpr int   N1  = 16384;
constexpr int   C_  = 128;
constexpr int   K1  = 1366;   // ceil(16384/12)
constexpr int   KP1 = 1376;   // padded to 16
constexpr int   K2  = 114;    // ceil(1366/12)
constexpr int   KP2 = 128;
constexpr int   N2  = 1366;   // level-2 row count (= K1)
constexpr int   NP2 = 1376;
constexpr float EMPTY_EPS = 1e-6f;

constexpr size_t A256(size_t x) { return (x + 255) & ~(size_t)255; }

// workspace layout (bytes)
constexpr size_t OFF_MU1   = 0;
constexpr size_t OFF_SIGM1 = A256(OFF_MU1   + (size_t)B_*N1*3*4);
constexpr size_t OFF_SBF   = A256(OFF_SIGM1 + (size_t)B_*N1*9*4);
constexpr size_t OFF_STBF  = A256(OFF_SBF   + (size_t)B_*N1*C_*2);
constexpr size_t OFF_W0T   = A256(OFF_STBF  + (size_t)B_*N1*C_*2);
constexpr size_t OFF_W1T   = A256(OFF_W0T   + (size_t)C_*C_*2);
constexpr size_t OFF_Q1    = A256(OFF_W1T   + (size_t)C_*C_*2);
constexpr size_t OFF_C1    = A256(OFF_Q1    + (size_t)B_*N1*C_*2);
constexpr size_t OFF_AT1   = A256(OFF_C1    + (size_t)B_*KP1*C_*2);
constexpr size_t OFF_CNT1  = A256(OFF_AT1   + (size_t)B_*KP1*N1*2);
constexpr size_t OFF_MCW1  = A256(OFF_CNT1  + (size_t)B_*KP1*4);
constexpr size_t OFF_MUC1  = A256(OFF_MCW1  + (size_t)B_*KP1*4);
constexpr size_t OFF_SGC1  = A256(OFF_MUC1  + (size_t)B_*KP1*3*4);
constexpr size_t OFF_SIGM2 = A256(OFF_SGC1  + (size_t)B_*KP1*9*4);
constexpr size_t OFF_SCBF  = A256(OFF_SIGM2 + (size_t)B_*KP1*9*4);
constexpr size_t OFF_ST2   = A256(OFF_SCBF  + (size_t)B_*KP1*C_*2);
constexpr size_t OFF_Q2    = A256(OFF_ST2   + (size_t)B_*C_*NP2*2);
constexpr size_t OFF_C2    = A256(OFF_Q2    + (size_t)B_*KP1*C_*2);
constexpr size_t OFF_AT2   = A256(OFF_C2    + (size_t)B_*KP2*C_*2);
constexpr size_t OFF_CNT2  = A256(OFF_AT2   + (size_t)B_*KP2*NP2*2);
constexpr size_t OFF_MCW2  = A256(OFF_CNT2  + (size_t)B_*KP2*4);
constexpr size_t OFF_MUC2  = A256(OFF_MCW2  + (size_t)B_*KP2*4);
constexpr size_t OFF_SGC2  = A256(OFF_MUC2  + (size_t)B_*KP2*3*4);
constexpr size_t OFF_REP1  = A256(OFF_SGC2  + (size_t)B_*KP2*9*4);
constexpr size_t OFF_REP2  = A256(OFF_REP1  + (size_t)B_*KP1*4);
constexpr size_t WS_TOTAL  = A256(OFF_REP2  + (size_t)B_*KP2*4);

// output layout (f32 elements, concatenated in return order)
constexpr size_t O_SC0 = 0;
constexpr size_t O_A0  = O_SC0 + (size_t)B_*K1*C_;
constexpr size_t O_MC0 = O_A0  + (size_t)B_*N1*K1;
constexpr size_t O_SC1 = O_MC0 + (size_t)B_*K1;
constexpr size_t O_A1  = O_SC1 + (size_t)B_*K2*C_;
constexpr size_t O_MC1 = O_A1  + (size_t)B_*N2*K2;
constexpr size_t O_REG = O_MC1 + (size_t)B_*K2;

// ---------------------------------------------------------------- WMMA types
typedef __bf16 bf16;
typedef __attribute__((ext_vector_type(16))) __bf16 v16bf;
typedef __attribute__((ext_vector_type(8)))  float  v8f;

__device__ __forceinline__ v8f wmma_bf16(v16bf a, v16bf b, v8f c) {
  // D(16x16,f32) = A(16x32,bf16) x B(32x16,bf16) + C
  return __builtin_amdgcn_wmma_f32_16x16x32_bf16(false, a, false, b, (short)0, c,
                                                 false, false);
}

// A-matrix 16x32 bf16 fragment from row-major [16 x ld] (contraction contiguous)
__device__ __forceinline__ v16bf load_a_frag(const bf16* __restrict__ base, int ld,
                                             int lane) {
  const int row = lane & 15;
  const int hf  = lane >> 4;               // 0 or 1
  const bf16* p0 = base + (size_t)row * ld + hf * 16;
  const bf16* p1 = p0 + 8;
  v16bf f;
#pragma unroll
  for (int i = 0; i < 8; ++i) { f[i] = p0[i]; f[8 + i] = p1[i]; }
  return f;
}

// B-matrix 32x16 bf16 fragment; memory holds columns contiguously: mem[col][k]
__device__ __forceinline__ v16bf load_b_frag(const bf16* __restrict__ base, int ld,
                                             int lane) {
  const int col = lane & 15;
  const int kh  = (lane >> 4) << 4;        // 0 or 16
  const bf16* p = base + (size_t)col * ld + kh;
  v16bf f;
#pragma unroll
  for (int i = 0; i < 16; ++i) f[i] = p[i];
  return f;
}

// ---------------------------------------------------------------- utility kernels
__global__ void zero_kernel(uint32_t* p, size_t nwords) {
  size_t i = (size_t)blockIdx.x * blockDim.x + threadIdx.x;
  if (i < nwords) p[i] = 0u;
}

// mu = trans + rot@local_mean ; SigM = rot diag(exp(2 slog)) rotT + mu muT
__global__ void prep_kernel(const float* __restrict__ rot, const float* __restrict__ trans,
                            const float* __restrict__ slog, const float* __restrict__ lmean,
                            float* __restrict__ mu, float* __restrict__ sigm) {
  int t = blockIdx.x * 256 + threadIdx.x;
  if (t >= B_ * N1) return;
  const float* R  = rot + (size_t)t * 9;
  const float* lm = lmean + (size_t)t * 3;
  const float* tr = trans + (size_t)t * 3;
  const float* sl = slog + (size_t)t * 3;
  float m[3], var[3];
#pragma unroll
  for (int i = 0; i < 3; ++i)
    m[i] = tr[i] + R[3*i]*lm[0] + R[3*i+1]*lm[1] + R[3*i+2]*lm[2];
#pragma unroll
  for (int j = 0; j < 3; ++j) var[j] = __expf(2.0f * sl[j]);
#pragma unroll
  for (int i = 0; i < 3; ++i)
#pragma unroll
    for (int j = 0; j < 3; ++j) {
      float s = R[3*i]*var[0]*R[3*j] + R[3*i+1]*var[1]*R[3*j+1] + R[3*i+2]*var[2]*R[3*j+2];
      sigm[(size_t)t*9 + 3*i + j] = s + m[i]*m[j];
    }
#pragma unroll
  for (int i = 0; i < 3; ++i) mu[(size_t)t*3 + i] = m[i];
}

// s (f32) -> s_bf [b][n][c] and st_bf [b][c][n]
__global__ void cast_s_kernel(const float* __restrict__ s, bf16* __restrict__ sbf,
                              bf16* __restrict__ st) {
  size_t t = (size_t)blockIdx.x * 256 + threadIdx.x;
  if (t >= (size_t)B_*N1*C_) return;
  int c = (int)(t % C_);
  size_t bn = t / C_;
  int n = (int)(bn % N1);
  int b = (int)(bn / N1);
  bf16 v = (bf16)s[t];
  sbf[t] = v;
  st[((size_t)b*C_ + c)*N1 + n] = v;
}

// W0,W1 (f32 [k][j]) -> transposed bf16 [j][k]
__global__ void cast_w_kernel(const float* __restrict__ W0, const float* __restrict__ W1,
                              bf16* __restrict__ W0t, bf16* __restrict__ W1t) {
  int t = blockIdx.x * 256 + threadIdx.x;
  if (t >= 2 * C_ * C_) return;
  int which = t / (C_*C_);
  int e = t % (C_*C_);
  int r = e / C_, c = e % C_;
  const float* W = which ? W1 : W0;
  bf16* Wt = which ? W1t : W0t;
  Wt[(size_t)c*C_ + r] = (bf16)W[e];
}

// ---------------------------------------------------------------- GEMM: q = s @ W
__global__ void gemm_q_kernel(const bf16* __restrict__ S, const bf16* __restrict__ Wt,
                              bf16* __restrict__ Q, int Mpad) {
  int b    = blockIdx.y;
  int wid  = blockIdx.x * 8 + (threadIdx.x >> 5);
  int lane = threadIdx.x & 31;
  int tm = wid >> 3, tn = wid & 7;
  if (tm * 16 >= Mpad) return;
  const bf16* Sb = S + ((size_t)b * Mpad + (size_t)tm * 16) * C_;
  v8f acc = {};
#pragma unroll
  for (int kk = 0; kk < C_; kk += 32) {
    v16bf a  = load_a_frag(Sb + kk, C_, lane);
    v16bf bb = load_b_frag(Wt + (size_t)tn * 16 * C_ + kk, C_, lane);
    acc = wmma_bf16(a, bb, acc);
  }
  bf16* Qb = Q + (size_t)b * Mpad * C_;
  int hf = lane >> 4, col = tn * 16 + (lane & 15);
#pragma unroll
  for (int r = 0; r < 8; ++r)
    Qb[((size_t)tm * 16 + r + 8 * hf) * C_ + col] = (bf16)acc[r];
}

// centers[b][k][c] = q[b][k*Nsrc/Kreal][c]
__global__ void gather_centers_kernel(const bf16* __restrict__ Q, bf16* __restrict__ Cent,
                                      int Mpad, int Nsrc, int Kreal, int Kpad) {
  size_t t = (size_t)blockIdx.x * 256 + threadIdx.x;
  if (t >= (size_t)B_ * Kreal * C_) return;
  int c = (int)(t % C_);
  size_t bk = t / C_;
  int k = (int)(bk % Kreal);
  int b = (int)(bk / Kreal);
  long idx = ((long)k * Nsrc) / Kreal;
  Cent[((size_t)b * Kpad + k) * C_ + c] = Q[((size_t)b * Mpad + idx) * C_ + c];
}

// ---------------------------------------------------------------- fused logits+softmax
// one workgroup = 16 rows; 8 waves split K-tiles; logits staged in dynamic LDS.
__global__ void logits_softmax_kernel(const bf16* __restrict__ Q, const bf16* __restrict__ Cent,
                                      const float* __restrict__ mask, int mask_ld,
                                      float* __restrict__ Aout, bf16* __restrict__ At,
                                      int Mpad, int Nvalid, int Kreal, int Kpad,
                                      int Npad_at, float scale) {
  extern __shared__ float lds[];           // [16][Kpad]
  __shared__ float red[16][17];
  __shared__ float rowstat[16];
  int b    = blockIdx.y;
  int n0   = blockIdx.x * 16;
  int lane = threadIdx.x & 31;
  int wave = threadIdx.x >> 5;

  const bf16* Qb = Q + ((size_t)b * Mpad + n0) * C_;
  v16bf af[4];
#pragma unroll
  for (int kk = 0; kk < 4; ++kk) af[kk] = load_a_frag(Qb + kk * 32, C_, lane);

  int ktiles = Kpad >> 4;
  for (int kt = wave; kt < ktiles; kt += 8) {
    const bf16* Cb = Cent + ((size_t)b * Kpad + (size_t)kt * 16) * C_;
    v8f acc = {};
#pragma unroll
    for (int kk = 0; kk < 4; ++kk)
      acc = wmma_bf16(af[kk], load_b_frag(Cb + kk * 32, C_, lane), acc);
    int hf = lane >> 4, col = kt * 16 + (lane & 15);
#pragma unroll
    for (int r = 0; r < 8; ++r)
      lds[(size_t)(r + 8 * hf) * Kpad + col] = acc[r] * scale;
  }
  __syncthreads();

  int r = threadIdx.x >> 4, l16 = threadIdx.x & 15;
  float* row = lds + (size_t)r * Kpad;
  float mx = -3.0e38f;
  for (int k = l16; k < Kreal; k += 16) mx = fmaxf(mx, row[k]);
  red[r][l16] = mx;
  __syncthreads();
  if (l16 == 0) {
    float m2 = red[r][0];
#pragma unroll
    for (int i = 1; i < 16; ++i) m2 = fmaxf(m2, red[r][i]);
    rowstat[r] = m2;
  }
  __syncthreads();
  float rm = rowstat[r];
  float sm = 0.f;
  for (int k = l16; k < Kreal; k += 16) { float e = __expf(row[k] - rm); row[k] = e; sm += e; }
  red[r][l16] = sm;
  __syncthreads();
  if (l16 == 0) {
    float s2 = 0.f;
#pragma unroll
    for (int i = 0; i < 16; ++i) s2 += red[r][i];
    rowstat[r] = s2;
  }
  __syncthreads();
  int nrow = n0 + r;
  float mval = (nrow < Nvalid) ? mask[(size_t)b * mask_ld + nrow] : 0.f;
  float inv  = mval / rowstat[r];
  float* arow = Aout + ((size_t)b * Nvalid + nrow) * Kreal;
  for (int k = l16; k < Kreal; k += 16) {
    float a = row[k] * inv;
    row[k] = a;
    if (nrow < Nvalid) arow[k] = a;
  }
  __syncthreads();
  // write transposed bf16 A for the s_c WMMA (contiguous n per k row)
  for (int k = threadIdx.x; k < Kreal; k += 256) {
    bf16* atp = At + ((size_t)b * Kpad + k) * Npad_at + n0;
#pragma unroll
    for (int rr = 0; rr < 16; ++rr) atp[rr] = (bf16)lds[(size_t)rr * Kpad + k];
  }
}

// counts[b][k] = sum_n A[b][n][k]  (deterministic per-(b,k) wave reduction, f32 exact)
__global__ void counts_kernel(const float* __restrict__ Aout, float* __restrict__ counts,
                              int Nvalid, int Kreal, int Kpad) {
  int wid  = blockIdx.x * 8 + (threadIdx.x >> 5);
  int lane = threadIdx.x & 31;
  if (wid >= B_ * Kreal) return;
  int b = wid / Kreal, k = wid % Kreal;
  const float* ap = Aout + (size_t)b * Nvalid * Kreal + k;
  float s = 0.f;
  for (int n = lane; n < Nvalid; n += 32) s += ap[(size_t)n * Kreal];
#pragma unroll
  for (int off = 16; off > 0; off >>= 1) s += __shfl_down(s, off, 32);
  if (lane == 0) counts[(size_t)b * Kpad + k] = s;
}

__global__ void mcounts_kernel(const float* __restrict__ counts, float* __restrict__ mc_out,
                               float* __restrict__ mcws, int Kreal, int Kpad) {
  int t = blockIdx.x * 256 + threadIdx.x;
  if (t >= B_ * Kreal) return;
  int b = t / Kreal, k = t % Kreal;
  float v = (counts[(size_t)b * Kpad + k] > EMPTY_EPS) ? 1.f : 0.f;
  mc_out[(size_t)b * Kreal + k] = v;
  mcws[(size_t)b * Kpad + k]   = v;
}

// ---------------------------------------------------------------- s_c = (A^T @ s)/counts * m_c
__global__ void sc_gemm_kernel(const bf16* __restrict__ At, const bf16* __restrict__ St,
                               const float* __restrict__ counts, const float* __restrict__ mcws,
                               float* __restrict__ out_sc, bf16* __restrict__ scbf,
                               bf16* __restrict__ st2, int st2ld,
                               int Kpad, int Kreal, int Npad) {
  int b    = blockIdx.y;
  int wid  = blockIdx.x * 8 + (threadIdx.x >> 5);
  int lane = threadIdx.x & 31;
  int tm = wid >> 3, tn = wid & 7;
  const bf16* Ab = At + ((size_t)b * Kpad + (size_t)tm * 16) * Npad;
  const bf16* Sb = St + ((size_t)b * C_  + (size_t)tn * 16) * Npad;
  v8f acc = {};
  for (int n = 0; n < Npad; n += 32) {
    v16bf a  = load_a_frag(Ab + n, Npad, lane);
    v16bf bb = load_b_frag(Sb + n, Npad, lane);
    acc = wmma_bf16(a, bb, acc);
  }
  int hf = lane >> 4, col = tn * 16 + (lane & 15);
#pragma unroll
  for (int r = 0; r < 8; ++r) {
    int k = tm * 16 + r + 8 * hf;
    if (k < Kreal) {
      float cnt = counts[(size_t)b * Kpad + k];
      float v = acc[r] / (cnt + EMPTY_EPS) * mcws[(size_t)b * Kpad + k];
      out_sc[((size_t)b * Kreal + k) * C_ + col] = v;
      if (scbf) scbf[((size_t)b * Kpad + k) * C_ + col] = (bf16)v;
      if (st2)  st2[((size_t)b * C_ + col) * st2ld + k] = (bf16)v;
    }
  }
}

// ---------------------------------------------------------------- mu_c / Sigma_c reduction
__global__ void musig_kernel(const bf16* __restrict__ At, const float* __restrict__ muv,
                             const float* __restrict__ sigm, const float* __restrict__ counts,
                             float* __restrict__ mu_c, float* __restrict__ sig_c,
                             int Kpad, int Kreal, int Npad_at, int Nvalid, int nb) {
  int wid  = blockIdx.x * 8 + (threadIdx.x >> 5);
  int lane = threadIdx.x & 31;
  if (wid >= B_ * Kreal) return;
  int b = wid / Kreal, k = wid % Kreal;
  const bf16* ap = At + ((size_t)b * Kpad + k) * Npad_at;
  const float* mp = muv  + (size_t)b * nb * 3;
  const float* sp = sigm + (size_t)b * nb * 9;
  float acc[12];
#pragma unroll
  for (int i = 0; i < 12; ++i) acc[i] = 0.f;
  for (int n = lane; n < Nvalid; n += 32) {
    float a = (float)ap[n];
    const float* mm = mp + (size_t)n * 3;
    const float* ss = sp + (size_t)n * 9;
    acc[0] += a * mm[0]; acc[1] += a * mm[1]; acc[2] += a * mm[2];
#pragma unroll
    for (int j = 0; j < 9; ++j) acc[3 + j] += a * ss[j];
  }
#pragma unroll
  for (int off = 16; off > 0; off >>= 1)
#pragma unroll
    for (int i = 0; i < 12; ++i) acc[i] += __shfl_down(acc[i], off, 32);
  if (lane == 0) {
    float c = counts[(size_t)b * Kpad + k] + EMPTY_EPS;
    float m0 = acc[0] / c, m1 = acc[1] / c, m2 = acc[2] / c;
    float mo[3] = {m0, m1, m2};
    mu_c[((size_t)b * Kpad + k) * 3 + 0] = m0;
    mu_c[((size_t)b * Kpad + k) * 3 + 1] = m1;
    mu_c[((size_t)b * Kpad + k) * 3 + 2] = m2;
#pragma unroll
    for (int i = 0; i < 3; ++i)
#pragma unroll
      for (int j = 0; j < 3; ++j)
        sig_c[((size_t)b * Kpad + k) * 9 + 3*i + j] = acc[3 + 3*i + j] / c - mo[i]*mo[j];
  }
}

// ---------------------------------------------------------------- coarse rigids -> Sigma(next) + mu muT
__global__ void rigid_kernel(const float* __restrict__ sig_c, const float* __restrict__ mu_c,
                             float* __restrict__ sigm2, int Kpad, int Kreal) {
  int t = blockIdx.x * 256 + threadIdx.x;
  if (t >= B_ * Kreal) return;
  int b = t / Kreal, k = t % Kreal;
  const float* S = sig_c + ((size_t)b * Kpad + k) * 9;
  float a[3][3], V[3][3] = {{1,0,0},{0,1,0},{0,0,1}};
#pragma unroll
  for (int i = 0; i < 3; ++i)
#pragma unroll
    for (int j = 0; j < 3; ++j) a[i][j] = 0.5f * (S[3*i + j] + S[3*j + i]);
#pragma unroll
  for (int i = 0; i < 3; ++i) a[i][i] += 1e-6f;  // SIGMA_JITTER
  // cyclic Jacobi, 6 sweeps
  for (int sweep = 0; sweep < 6; ++sweep) {
#pragma unroll
    for (int pair = 0; pair < 3; ++pair) {
      int p = (pair == 2) ? 1 : 0;
      int q = (pair == 0) ? 1 : 2;
      float apq = a[p][q];
      if (fabsf(apq) > 1e-14f) {
        float theta = (a[q][q] - a[p][p]) / (2.0f * apq);
        float tt = copysignf(1.0f, theta) / (fabsf(theta) + sqrtf(theta*theta + 1.0f));
        float cc = rsqrtf(tt*tt + 1.0f);
        float ss = tt * cc;
        float app = a[p][p], aqq = a[q][q];
        a[p][p] = cc*cc*app - 2.f*ss*cc*apq + ss*ss*aqq;
        a[q][q] = ss*ss*app + 2.f*ss*cc*apq + cc*cc*aqq;
        a[p][q] = a[q][p] = 0.f;
        int rr = 3 - p - q;
        float arp = a[rr][p], arq = a[rr][q];
        a[rr][p] = a[p][rr] = cc*arp - ss*arq;
        a[rr][q] = a[q][rr] = ss*arp + cc*arq;
#pragma unroll
        for (int i = 0; i < 3; ++i) {
          float vp = V[i][p], vq = V[i][q];
          V[i][p] = cc*vp - ss*vq;
          V[i][q] = ss*vp + cc*vq;
        }
      }
    }
  }
  float lam[3];
#pragma unroll
  for (int i = 0; i < 3; ++i) {
    float l  = fmaxf(a[i][i], 1e-6f);           // VAR_FLOOR
    float sc = fmaxf(sqrtf(l), 1e-6f);
    sc = sc + 1e-6f;                             // exp(2*log(scaling+1e-6))
    lam[i] = sc * sc;
  }
  const float* mu = mu_c + ((size_t)b * Kpad + k) * 3;
#pragma unroll
  for (int i = 0; i < 3; ++i)
#pragma unroll
    for (int j = 0; j < 3; ++j) {
      float s = V[i][0]*lam[0]*V[j][0] + V[i][1]*lam[1]*V[j][1] + V[i][2]*lam[2]*V[j][2];
      sigm2[((size_t)b * Kpad + k) * 9 + 3*i + j] = s + mu[i]*mu[j];
    }
}

// ---------------------------------------------------------------- repulsion per-center sums
__global__ void rep_kernel(const float* __restrict__ mu_c, float* __restrict__ rep_out,
                           int Kpad, int Kreal) {
  int t = blockIdx.x * 256 + threadIdx.x;
  if (t >= B_ * Kreal) return;
  int b = t / Kreal, i = t % Kreal;
  const float* mb = mu_c + (size_t)b * Kpad * 3;
  float x = mb[(size_t)i*3], y = mb[(size_t)i*3+1], z = mb[(size_t)i*3+2];
  float nn[4] = {1e30f, 1e30f, 1e30f, 1e30f};
  for (int j = 0; j < Kreal; ++j) {
    float dx = x - mb[(size_t)j*3], dy = y - mb[(size_t)j*3+1], dz = z - mb[(size_t)j*3+2];
    float d2 = dx*dx + dy*dy + dz*dz;
    if (j == i) d2 += 1e9f;
    if (d2 < nn[3]) {
      nn[3] = d2;
#pragma unroll
      for (int q = 3; q > 0; --q)
        if (nn[q] < nn[q-1]) { float tmp = nn[q]; nn[q] = nn[q-1]; nn[q-1] = tmp; }
    }
  }
  float s = 0.f;
#pragma unroll
  for (int q = 0; q < 4; ++q) s += __expf(-sqrtf(nn[q] + 1e-12f));
  rep_out[t] = s;
}

// ---------------------------------------------------------------- deterministic reg finalize
__global__ void final_reg_kernel(const float* __restrict__ cnt1, const float* __restrict__ cnt2,
                                 const float* __restrict__ rep1, const float* __restrict__ rep2,
                                 float* __restrict__ out) {
  __shared__ float sred[256];
  const int tid = threadIdx.x;
  auto breduce = [&](float v) -> float {
    sred[tid] = v; __syncthreads();
    for (int s = 128; s > 0; s >>= 1) {
      if (tid < s) sred[tid] += sred[tid + s];
      __syncthreads();
    }
    float r = sred[0]; __syncthreads();
    return r;
  };
  float reg = 0.f;
  for (int lvl = 0; lvl < 2; ++lvl) {
    const float* cnt = lvl ? cnt2 : cnt1;
    const float* rep = lvl ? rep2 : rep1;
    int Kreal = lvl ? K2 : K1;
    int Kpad  = lvl ? KP2 : KP1;
    float occ = 0.f, col = 0.f;
    for (int b = 0; b < B_; ++b) {
      float tsum = 0.f;
      for (int k = tid; k < Kreal; k += 256) tsum += cnt[(size_t)b * Kpad + k];
      float tot = breduce(tsum) + EMPTY_EPS;
      float pp = 0.f, ce = 0.f;
      for (int k = tid; k < Kreal; k += 256) {
        float c = cnt[(size_t)b * Kpad + k];
        float p = c / tot;
        pp += p * p;
        ce += __expf(-c);
      }
      occ += (float)Kreal * breduce(pp);
      col += breduce(ce);
    }
    float rs = 0.f;
    for (int t2 = tid; t2 < B_ * Kreal; t2 += 256) rs += rep[t2];
    float rsum = breduce(rs);
    reg += 1.0f * (occ / (float)B_)                       // W_OCC * occ
         + 0.1f * (rsum / ((float)B_ * Kreal * 4.0f))     // W_REP * rep
         + (col / ((float)B_ * Kreal));                   // collapse
  }
  if (tid == 0) out[0] = reg;
}

// ---------------------------------------------------------------- host launcher
static inline int ceildiv(long a, long b) { return (int)((a + b - 1) / b); }

extern "C" void kernel_launch(void* const* d_in, const int* in_sizes, int n_in,
                              void* d_out, int out_size, void* d_ws, size_t ws_size,
                              hipStream_t stream) {
  (void)in_sizes; (void)n_in; (void)out_size; (void)ws_size;
  const float* s_f   = (const float*)d_in[0];
  const float* rot   = (const float*)d_in[1];
  const float* trans = (const float*)d_in[2];
  const float* slog  = (const float*)d_in[3];
  const float* lmean = (const float*)d_in[4];
  const float* mask  = (const float*)d_in[5];
  const float* W0    = (const float*)d_in[6];
  const float* W1    = (const float*)d_in[7];
  float* out = (float*)d_out;
  char*  ws  = (char*)d_ws;

  float* mu1   = (float*)(ws + OFF_MU1);
  float* sigm1 = (float*)(ws + OFF_SIGM1);
  bf16*  sbf   = (bf16*)(ws + OFF_SBF);
  bf16*  stbf  = (bf16*)(ws + OFF_STBF);
  bf16*  w0t   = (bf16*)(ws + OFF_W0T);
  bf16*  w1t   = (bf16*)(ws + OFF_W1T);
  bf16*  q1    = (bf16*)(ws + OFF_Q1);
  bf16*  c1    = (bf16*)(ws + OFF_C1);
  bf16*  at1   = (bf16*)(ws + OFF_AT1);
  float* cnt1  = (float*)(ws + OFF_CNT1);
  float* mcw1  = (float*)(ws + OFF_MCW1);
  float* muc1  = (float*)(ws + OFF_MUC1);
  float* sgc1  = (float*)(ws + OFF_SGC1);
  float* sigm2 = (float*)(ws + OFF_SIGM2);
  bf16*  scbf  = (bf16*)(ws + OFF_SCBF);
  bf16*  st2   = (bf16*)(ws + OFF_ST2);
  bf16*  q2    = (bf16*)(ws + OFF_Q2);
  bf16*  c2    = (bf16*)(ws + OFF_C2);
  bf16*  at2   = (bf16*)(ws + OFF_AT2);
  float* cnt2  = (float*)(ws + OFF_CNT2);
  float* mcw2  = (float*)(ws + OFF_MCW2);
  float* muc2  = (float*)(ws + OFF_MUC2);
  float* sgc2  = (float*)(ws + OFF_SGC2);
  float* rep1  = (float*)(ws + OFF_REP1);
  float* rep2  = (float*)(ws + OFF_REP2);

  const float scale = 1.0f / (sqrtf((float)C_) * 2.0f);  // 1/(sqrt(C)*TAU)

  // 0) zero workspace (pads, counters)
  {
    size_t words = WS_TOTAL / 4;
    zero_kernel<<<ceildiv((long)words, 256), 256, 0, stream>>>((uint32_t*)ws, words);
  }
  // 1) elementwise prep + casts
  prep_kernel<<<ceildiv((long)B_*N1, 256), 256, 0, stream>>>(rot, trans, slog, lmean, mu1, sigm1);
  cast_s_kernel<<<ceildiv((long)B_*N1*C_, 256), 256, 0, stream>>>(s_f, sbf, stbf);
  cast_w_kernel<<<ceildiv((long)2*C_*C_, 256), 256, 0, stream>>>(W0, W1, w0t, w1t);

  // ---------------- level 1 ----------------
  gemm_q_kernel<<<dim3(N1/16, B_), 256, 0, stream>>>(sbf, w0t, q1, N1);
  gather_centers_kernel<<<ceildiv((long)B_*K1*C_, 256), 256, 0, stream>>>(q1, c1, N1, N1, K1, KP1);
  logits_softmax_kernel<<<dim3(N1/16, B_), 256, (size_t)16*KP1*4, stream>>>(
      q1, c1, mask, N1, out + O_A0, at1, N1, N1, K1, KP1, N1, scale);
  counts_kernel<<<ceildiv((long)B_*K1, 8), 256, 0, stream>>>(out + O_A0, cnt1, N1, K1, KP1);
  mcounts_kernel<<<ceildiv((long)B_*K1, 256), 256, 0, stream>>>(cnt1, out + O_MC0, mcw1, K1, KP1);
  sc_gemm_kernel<<<dim3(KP1/16, B_), 256, 0, stream>>>(
      at1, stbf, cnt1, mcw1, out + O_SC0, scbf, st2, NP2, KP1, K1, N1);
  musig_kernel<<<ceildiv((long)B_*K1, 8), 256, 0, stream>>>(
      at1, mu1, sigm1, cnt1, muc1, sgc1, KP1, K1, N1, N1, N1);
  rep_kernel<<<ceildiv((long)B_*K1, 256), 256, 0, stream>>>(muc1, rep1, KP1, K1);
  rigid_kernel<<<ceildiv((long)B_*K1, 256), 256, 0, stream>>>(sgc1, muc1, sigm2, KP1, K1);

  // ---------------- level 2 ----------------
  gemm_q_kernel<<<dim3(KP1/16, B_), 256, 0, stream>>>(scbf, w1t, q2, KP1);
  gather_centers_kernel<<<ceildiv((long)B_*K2*C_, 256), 256, 0, stream>>>(q2, c2, KP1, N2, K2, KP2);
  logits_softmax_kernel<<<dim3(NP2/16, B_), 256, (size_t)16*KP2*4, stream>>>(
      q2, c2, mcw1, KP1, out + O_A1, at2, KP1, N2, K2, KP2, NP2, scale);
  counts_kernel<<<ceildiv((long)B_*K2, 8), 256, 0, stream>>>(out + O_A1, cnt2, N2, K2, KP2);
  mcounts_kernel<<<ceildiv((long)B_*K2, 256), 256, 0, stream>>>(cnt2, out + O_MC1, mcw2, K2, KP2);
  sc_gemm_kernel<<<dim3(KP2/16, B_), 256, 0, stream>>>(
      at2, st2, cnt2, mcw2, out + O_SC1, (bf16*)nullptr, (bf16*)nullptr, 0, KP2, K2, NP2);
  musig_kernel<<<ceildiv((long)B_*K2, 8), 256, 0, stream>>>(
      at2, muc1, sigm2, cnt2, muc2, sgc2, KP2, K2, NP2, N2, KP1);
  rep_kernel<<<ceildiv((long)B_*K2, 256), 256, 0, stream>>>(muc2, rep2, KP2, K2);

  // ---------------- regularizer ----------------
  final_reg_kernel<<<1, 256, 0, stream>>>(cnt1, cnt2, rep1, rep2, out + O_REG);
}
